// Generator_24790551232894
// MI455X (gfx1250) — matrix operand order repaired and checked by reading, not compiled, not run on hardware
//
#include <hip/hip_runtime.h>
#include <hip/hip_bf16.h>

// ---- types ------------------------------------------------------------
typedef __attribute__((ext_vector_type(8)))  float  v8f;
typedef __attribute__((ext_vector_type(16))) __bf16 v16bf;
typedef __attribute__((ext_vector_type(8)))  __bf16 v8bf;

#define Hp    320          // padded hidden (300 -> 320), K multiple of 32
#define GATES 1280         // 4 * Hp
#define BT    32           // batch rows per workgroup
#define NTHREADS 256       // 8 waves of 32
#define NGROUPS  5         // (BT/16 rowtiles) * (Hp/16 coltiles) / 8 waves
#define BUFE  (BT * Hp)    // 10240 elements per h/c buffer
#define NWG   128          // 4096 / BT

// ---- fast activations -------------------------------------------------
__device__ __forceinline__ float sigf(float x) {
    return 1.0f / (1.0f + __expf(-x));
}
__device__ __forceinline__ float tanh_fast(float x) {
    return 1.0f - 2.0f / (__expf(2.0f * x) + 1.0f);
}

// ---- A-operand assemble from two 16B LDS reads ------------------------
__device__ __forceinline__ v16bf packA(const __bf16* p) {
    v8bf lo = *(const v8bf*)(p);
    v8bf hi = *(const v8bf*)(p + 8);
    v16bf a;
#pragma unroll
    for (int i = 0; i < 8; ++i) { a[i] = lo[i]; a[i + 8] = hi[i]; }
    return a;
}

// ---- load one k-step's 8 B tiles (4 gate quadrants x {Wx,Wh}) ---------
__device__ __forceinline__ void loadB(v16bf bx[4], v16bf bh[4],
                                      const __bf16* wxp, const __bf16* whp,
                                      int kb) {
    bx[0] = *(const v16bf*)(wxp + 0 * Hp * Hp + kb);
    bx[1] = *(const v16bf*)(wxp + 1 * Hp * Hp + kb);
    bx[2] = *(const v16bf*)(wxp + 2 * Hp * Hp + kb);
    bx[3] = *(const v16bf*)(wxp + 3 * Hp * Hp + kb);
    bh[0] = *(const v16bf*)(whp + 0 * Hp * Hp + kb);
    bh[1] = *(const v16bf*)(whp + 1 * Hp * Hp + kb);
    bh[2] = *(const v16bf*)(whp + 2 * Hp * Hp + kb);
    bh[3] = *(const v16bf*)(whp + 3 * Hp * Hp + kb);
}

// ---- one LSTM layer step for this WG's 32-row batch tile --------------
// gates[32,1280] = xsrc@Wx^T + hsrc@Wh^T + bias ; then elementwise update.
// Each wave owns NGROUPS (rowtile, hcoltile) groups and computes all 4 gate
// quadrants for that footprint, so i/f/g/o meet in the WMMA accumulators.
// c-state is per-thread private, kept in a global scratch slice (L2) to keep
// LDS small enough for 2 workgroups per WGP.
__device__ void lstm_phase(
    const __bf16* __restrict__ xsrc,   // LDS  [32][Hp] bf16
    const __bf16* __restrict__ hsrc,   // LDS  [32][Hp] bf16
    __bf16* __restrict__ hdst,         // LDS  [32][Hp] bf16
    float* __restrict__ cbuf,          // global [32][Hp] f32 (this layer's c)
    const __bf16* __restrict__ Wx,     // global [1280][Hp] bf16 (padded)
    const __bf16* __restrict__ Wh,     // global [1280][Hp] bf16 (padded)
    const float* __restrict__ bias,    // LDS [1280] f32 (b_ih+b_hh, padded)
    int wave, int lane,
    float* __restrict__ outp,          // null for layer 0
    int wgRow, int t, int T)
{
    const int hf = lane >> 4;
    const int ln = lane & 15;

#pragma unroll 1
    for (int gi = 0; gi < NGROUPS; ++gi) {
        const int g       = wave + gi * 8;      // 0..39
        const int rowbase = (g / 20) * 16;      // 0 or 16
        const int colbase = (g % 20) * 16;      // 0..304
        const int col     = colbase + ln;       // hidden column 0..319

        // Loop-invariant base pointers; quadrant strides and k-offsets fold
        // into the 24-bit instruction immediate.
        const __bf16* apx = xsrc + (rowbase + ln) * Hp + (hf ? 16 : 0);
        const __bf16* aph = hsrc + (rowbase + ln) * Hp + (hf ? 16 : 0);
        const __bf16* wxp = Wx + (size_t)col * Hp + hf * 16;
        const __bf16* whp = Wh + (size_t)col * Hp + hf * 16;

        v8f acc[4];
#pragma unroll
        for (int q = 0; q < 4; ++q) {
            const float bq = bias[q * Hp + col];
#pragma unroll
            for (int r = 0; r < 8; ++r) acc[q][r] = bq;
        }

        // K reduction: 10 steps of 32, fully unrolled, double-buffered B
        // tiles (scheduler may coalesce; inter-wave overlap at 4 waves/SIMD
        // is the primary latency cover).
        v16bf bxb[2][4], bhb[2][4];
        loadB(bxb[0], bhb[0], wxp, whp, 0);
#pragma unroll
        for (int ks = 0; ks < 10; ++ks) {
            const int cb = ks & 1;
            const int nb = cb ^ 1;
            if (ks < 9) {
                loadB(bxb[nb], bhb[nb], wxp, whp, (ks + 1) * 32);
            }
            const int kb = ks * 32;
            const v16bf ax = packA(apx + kb);
            const v16bf ah = packA(aph + kb);
#pragma unroll
            for (int q = 0; q < 4; ++q) {
                acc[q] = __builtin_amdgcn_wmma_f32_16x16x32_bf16(
                    false, ax, false, bxb[cb][q], (short)0, acc[q], false, false);
            }
#pragma unroll
            for (int q = 0; q < 4; ++q) {
                acc[q] = __builtin_amdgcn_wmma_f32_16x16x32_bf16(
                    false, ah, false, bhb[cb][q], (short)0, acc[q], false, false);
            }
        }

        // Elementwise LSTM cell update; C/D tile layout: VGPR r, lane L:
        // M = r + 8*(L>=16), N = L%16.
#pragma unroll
        for (int r = 0; r < 8; ++r) {
            const float iv = sigf(acc[0][r]);
            const float fv = sigf(acc[1][r]);
            const float gv = tanh_fast(acc[2][r]);
            const float ov = sigf(acc[3][r]);
            const int M = rowbase + r + (hf << 3);   // local batch row 0..31
            const int cidx = M * Hp + col;
            const float c = fv * cbuf[cidx] + iv * gv;
            cbuf[cidx] = c;
            const float h = ov * tanh_fast(c);
            hdst[cidx] = (__bf16)h;
            if (outp != nullptr && col < 300) {
                outp[((size_t)(wgRow + M) * T + t) * 300 + col] = h;
            }
        }
    }
}

// ---- main persistent kernel: one WG per 32 batch rows, loops over T ----
__global__ __launch_bounds__(NTHREADS, 2)
void lstm2_wmma_kernel(const float* __restrict__ z,
                       const __bf16* __restrict__ Wih0,
                       const __bf16* __restrict__ Whh0,
                       const __bf16* __restrict__ Wih1,
                       const __bf16* __restrict__ Whh1,
                       const float* __restrict__ biasg,  // [2][1280] global
                       float* __restrict__ cws,          // [NWG][2][BUFE] f32
                       const int* __restrict__ slen,
                       float* __restrict__ out)
{
    extern __shared__ __align__(16) char smem[];
    __bf16* zx  = (__bf16*)smem;               // [32][Hp]           20480 B
    __bf16* h0b = zx  + BUFE;                  // 2 x [32][Hp]       40960 B
    __bf16* h1b = h0b + 2 * BUFE;              // 2 x [32][Hp]       40960 B
    float*  bls = (float*)(h1b + 2 * BUFE);    // [2][1280] f32      10240 B
                                               // total             112640 B

    const int tid   = threadIdx.x;
    const int wave  = tid >> 5;
    const int lane  = tid & 31;
    const int wgRow = blockIdx.x * BT;

    // This WG's private c-state slice in global scratch (L2-resident).
    float* cg = cws + (size_t)blockIdx.x * (2 * BUFE);

    // Stage z (f32 -> bf16, pad cols 300..319 with zero).
    for (int idx = tid; idx < BUFE; idx += NTHREADS) {
        const int r = idx / Hp, k = idx % Hp;
        const float v = (k < 300) ? z[(size_t)(wgRow + r) * 300 + k] : 0.0f;
        zx[idx] = (__bf16)v;
    }
    // Zero h (LDS) and c (global scratch) state.
    for (int idx = tid; idx < 2 * BUFE; idx += NTHREADS) {
        h0b[idx] = (__bf16)0.0f;
        h1b[idx] = (__bf16)0.0f;
        cg[idx] = 0.0f;
    }
    // Stage combined bias.
    for (int idx = tid; idx < 2 * GATES; idx += NTHREADS) {
        bls[idx] = biasg[idx];
    }
    __syncthreads();

    const int T = slen[0];

    int cur = 0;
    for (int t = 0; t < T; ++t) {
        const int nxt = cur ^ 1;
        // layer 0: x = (t==0 ? z : h1_prev), h = h0_prev -> h0_new
        const __bf16* xsrc = (t == 0) ? zx : (h1b + cur * BUFE);
        lstm_phase(xsrc, h0b + cur * BUFE, h0b + nxt * BUFE, cg,
                   Wih0, Whh0, bls, wave, lane,
                   nullptr, wgRow, t, T);
        __syncthreads();
        // layer 1: x = h0_new, h = h1_prev -> h1_new (also the output ys[t])
        lstm_phase(h0b + nxt * BUFE, h1b + cur * BUFE, h1b + nxt * BUFE,
                   cg + BUFE, Wih1, Whh1, bls + GATES, wave, lane,
                   out, wgRow, t, T);
        __syncthreads();
        cur = nxt;
    }
}

// ---- prep: f32 weights [1200,300] -> padded bf16 [1280][320] ----------
__global__ void cvt_weight_kernel(const float* __restrict__ W, __bf16* __restrict__ Wb) {
    const int idx = blockIdx.x * blockDim.x + threadIdx.x;
    if (idx >= GATES * Hp) return;
    const int pr = idx / Hp, k = idx % Hp;
    const int q  = pr / Hp, n = pr % Hp;
    float v = 0.0f;
    if (n < 300 && k < 300) v = W[(size_t)(q * 300 + n) * 300 + k];
    Wb[idx] = (__bf16)v;
}

// ---- prep: combined padded biases [2][1280] ---------------------------
__global__ void cvt_bias_kernel(const float* __restrict__ bi0, const float* __restrict__ bh0,
                                const float* __restrict__ bi1, const float* __restrict__ bh1,
                                float* __restrict__ bias) {
    const int idx = blockIdx.x * blockDim.x + threadIdx.x;
    if (idx >= 2 * GATES) return;
    const int l = idx / GATES, pr = idx % GATES;
    const int q = pr / Hp, n = pr % Hp;
    const float* bi = l ? bi1 : bi0;
    const float* bh = l ? bh1 : bh0;
    bias[idx] = (n < 300) ? (bi[q * 300 + n] + bh[q * 300 + n]) : 0.0f;
}

extern "C" void kernel_launch(void* const* d_in, const int* in_sizes, int n_in,
                              void* d_out, int out_size, void* d_ws, size_t ws_size,
                              hipStream_t stream) {
    const float* z    = (const float*)d_in[0];
    const float* Wih0 = (const float*)d_in[1];
    const float* Whh0 = (const float*)d_in[2];
    const float* bih0 = (const float*)d_in[3];
    const float* bhh0 = (const float*)d_in[4];
    const float* Wih1 = (const float*)d_in[5];
    const float* Whh1 = (const float*)d_in[6];
    const float* bih1 = (const float*)d_in[7];
    const float* bhh1 = (const float*)d_in[8];
    const int*   slen = (const int*)d_in[9];
    float*       out  = (float*)d_out;

    // workspace: 4 bf16 weight matrices [1280][320], f32 bias [2][1280],
    // then per-WG c-state [NWG][2][BUFE] f32.
    const size_t WELEMS = (size_t)GATES * Hp;           // 409600
    __bf16* wb0 = (__bf16*)d_ws;                         // Wih0
    __bf16* wb1 = wb0 + WELEMS;                          // Whh0
    __bf16* wb2 = wb1 + WELEMS;                          // Wih1
    __bf16* wb3 = wb2 + WELEMS;                          // Whh1
    float*  bb  = (float*)(wb3 + WELEMS);                // bias [2*1280]
    float*  cws = bb + 2 * GATES;                        // c [128][2][10240]

    const int wblocks = (int)((WELEMS + 255) / 256);
    cvt_weight_kernel<<<wblocks, 256, 0, stream>>>(Wih0, wb0);
    cvt_weight_kernel<<<wblocks, 256, 0, stream>>>(Whh0, wb1);
    cvt_weight_kernel<<<wblocks, 256, 0, stream>>>(Wih1, wb2);
    cvt_weight_kernel<<<wblocks, 256, 0, stream>>>(Whh1, wb3);
    cvt_bias_kernel<<<(2 * GATES + 255) / 256, 256, 0, stream>>>(bih0, bhh0, bih1, bhh1, bb);

    const int B = 4096;
    // LDS: z(20480) + h0(40960) + h1(40960) + bias(10240) = 112640 B
    // -> 2 workgroups per WGP (4 waves/SIMD) for latency hiding.
    const size_t shmem = (size_t)(5 * BUFE) * sizeof(__bf16)
                       + (size_t)(2 * GATES) * sizeof(float);
    lstm2_wmma_kernel<<<B / BT, NTHREADS, shmem, stream>>>(
        z, wb0, wb1, wb2, wb3, bb, cws, slen, out);
}
